// Tacotron2Loss_18502719111777
// MI455X (gfx1250) — matrix operand order repaired and checked by reading, not compiled, not run on hardware
//
#include <hip/hip_runtime.h>
#include <hip/hip_bf16.h>
#include <math.h>

#define N_MEL 80
#define T_LEN 2000
#define B_SZ  64
#define L_TXT 200
#define K_MAIN 1984   // 62 * 32; tail covers K = 1984..1999

typedef __attribute__((ext_vector_type(16))) __bf16 v16bf;
typedef __attribute__((ext_vector_type(8)))  float  v8f;

// workspace accumulator slots (floats at start of d_ws)
enum {
    A_MEL1 = 0,  // sum (mel_out - tgt)^2
    A_MEL2,      // sum (postnet - tgt)^2
    A_WMSE,      // sum w[c]^2 (postnet - tgt)^2
    A_L1,        // sum |postnet - tgt|
    A_SPEC,      // sum ((dp - dt))^2 over t-diffs
    A_PERC,      // sum fw[c] (postnet - tgt)^2
    A_GRAM,      // sum (Sp - St)^2 raw dot-product diffs
    A_BCE,       // sum bce terms
    A_MONO,      // sum argmax violations
    A_MEAN,      // sum (mean_p - mean_t)^2 over (b,c)
    A_VAR,       // sum (var_p - var_t)^2 over (b,c)
    A_COUNT
};

__global__ void init_acc_kernel(float* acc) {
    if (threadIdx.x < A_COUNT) acc[threadIdx.x] = 0.0f;
}

// ---------------------------------------------------------------------------
// Fused single-pass kernel: one block per (b, c) row of length T_LEN.
// ---------------------------------------------------------------------------
__global__ void row_stats_kernel(const float* __restrict__ mo,
                                 const float* __restrict__ mp,
                                 const float* __restrict__ mt,
                                 float* __restrict__ acc) {
    const int row = blockIdx.x;          // b * N_MEL + c
    const int c   = row % N_MEL;
    const long base = (long)row * T_LEN;

    float w = 1.0f;
    if (c >= N_MEL / 4 && c < 3 * N_MEL / 4) w = 1.5f;
    else if (c >= 3 * N_MEL / 4)             w = 0.8f;
    const float w2 = w * w;
    const float fw = __expf(-(float)c * (1.0f / 20.0f));

    float s_mel1 = 0.f, s_mel2 = 0.f, s_wmse = 0.f, s_l1 = 0.f;
    float s_spec = 0.f, s_perc = 0.f;
    float sp = 0.f, sp2 = 0.f, st = 0.f, st2 = 0.f;

    for (int t = threadIdx.x; t < T_LEN; t += blockDim.x) {
        float a = mo[base + t];
        float p = mp[base + t];
        float g = mt[base + t];
        float d0 = a - g;
        float d  = p - g;
        s_mel1 += d0 * d0;
        s_mel2 += d * d;
        s_wmse += w2 * d * d;
        s_l1   += fabsf(d);
        s_perc += fw * d * d;
        sp += p; sp2 += p * p; st += g; st2 += g * g;
        if (t + 1 < T_LEN) {
            float p1 = mp[base + t + 1];
            float g1 = mt[base + t + 1];
            float dd = (p1 - p) - (g1 - g);
            s_spec += dd * dd;
        }
    }

    __shared__ float red[256];
    float vals[10] = { s_mel1, s_mel2, s_wmse, s_l1, s_spec, s_perc, sp, sp2, st, st2 };
    float tot[10];
    for (int q = 0; q < 10; ++q) {
        red[threadIdx.x] = vals[q];
        __syncthreads();
        for (int s = 128; s > 0; s >>= 1) {
            if (threadIdx.x < s) red[threadIdx.x] += red[threadIdx.x + s];
            __syncthreads();
        }
        if (threadIdx.x == 0) tot[q] = red[0];
        __syncthreads();
    }

    if (threadIdx.x == 0) {
        atomicAdd(&acc[A_MEL1], tot[0]);
        atomicAdd(&acc[A_MEL2], tot[1]);
        atomicAdd(&acc[A_WMSE], tot[2]);
        atomicAdd(&acc[A_L1],   tot[3]);
        atomicAdd(&acc[A_SPEC], tot[4]);
        atomicAdd(&acc[A_PERC], tot[5]);
        const float T = (float)T_LEN;
        float mean_p = tot[6] / T;
        float mean_t = tot[8] / T;
        float var_p  = (tot[7] - T * mean_p * mean_p) / (T - 1.0f);  // ddof=1
        float var_t  = (tot[9] - T * mean_t * mean_t) / (T - 1.0f);
        float dm = mean_p - mean_t;
        float dv = var_p - var_t;
        atomicAdd(&acc[A_MEAN], dm * dm);
        atomicAdd(&acc[A_VAR],  dv * dv);
    }
}

// ---------------------------------------------------------------------------
// Gram-matrix via v_wmma_f32_16x16x32_bf16.
// A layout:              lane = M%16, K = (lane/16)*16 + e
// B layout:              lane = N%16, K = (lane/16)*16 + e
// Main loop: 62 unguarded K=32 steps with aligned float4 (b128) loads.
// Tail (K=1984..2015): A loads K=1984..1999 in all lanes (upper lanes unused);
// B invalid only for lanes>=16 -> branch-free lane mask, no OOB addresses.
// ---------------------------------------------------------------------------
__device__ __forceinline__ void cvt8(const float4 a, const float4 b,
                                     v16bf& v, int off) {
    v[off + 0] = (__bf16)a.x; v[off + 1] = (__bf16)a.y;
    v[off + 2] = (__bf16)a.z; v[off + 3] = (__bf16)a.w;
    v[off + 4] = (__bf16)b.x; v[off + 5] = (__bf16)b.y;
    v[off + 6] = (__bf16)b.z; v[off + 7] = (__bf16)b.w;
}

__device__ __forceinline__ v8f gram_tile(const float* __restrict__ arow,
                                         const float* __restrict__ brow,
                                         int agrp, int bgrp, float bmask) {
    v8f c = {};
    for (int k = 0; k < K_MAIN; k += 32) {
        const float4* ap0 = (const float4*)(arow + k + bgrp);
        const float4* ap1 = (const float4*)(arow + k + bgrp + 8);
        float4 a0 = ap0[0], a1 = ap0[1];
        float4 a2 = ap1[0], a3 = ap1[1];
        v16bf av;
        cvt8(a0, a1, av, 0);
        cvt8(a2, a3, av, 8);

        const float4* bp = (const float4*)(brow + k + bgrp);
        float4 b0 = bp[0], b1 = bp[1], b2 = bp[2], b3 = bp[3];
        v16bf bv;
        cvt8(b0, b1, bv, 0);
        cvt8(b2, b3, bv, 8);

        c = __builtin_amdgcn_wmma_f32_16x16x32_bf16(
                false, av, false, bv, (short)0, c, false, false);
    }
    // ---- tail step: K base = 1984, only K < 2000 valid ----
    {
        const float4* ap = (const float4*)(arow + K_MAIN); // K=1984..1999, valid
        float4 a0 = ap[0], a1 = ap[1], a2 = ap[2], a3 = ap[3];
        v16bf av;                      // lanes>=16 meet masked-zero B
        cvt8(a0, a1, av, 0);
        cvt8(a2, a3, av, 8);

        const float4* bp = (const float4*)(brow + K_MAIN);        // valid region only
        float4 b0 = bp[0], b1 = bp[1], b2 = bp[2], b3 = bp[3];
        b0.x *= bmask; b0.y *= bmask; b0.z *= bmask; b0.w *= bmask;
        b1.x *= bmask; b1.y *= bmask; b1.z *= bmask; b1.w *= bmask;
        b2.x *= bmask; b2.y *= bmask; b2.z *= bmask; b2.w *= bmask;
        b3.x *= bmask; b3.y *= bmask; b3.z *= bmask; b3.w *= bmask;
        v16bf bv;
        cvt8(b0, b1, bv, 0);
        cvt8(b2, b3, bv, 8);

        c = __builtin_amdgcn_wmma_f32_16x16x32_bf16(
                false, av, false, bv, (short)0, c, false, false);
    }
    return c;
}

__global__ void gram_wmma_kernel(const float* __restrict__ mp,
                                 const float* __restrict__ mt,
                                 float* __restrict__ acc) {
    const int b    = blockIdx.x;
    const int wave = threadIdx.x >> 5;
    const int lane = threadIdx.x & 31;
    const long pbase = (long)b * N_MEL * T_LEN;

    const int   lrow  = lane & 15;
    const int   agrp  = (lane >> 4) * 8;     // unused; A and B both offset K by bgrp
    const int   bgrp  = (lane >> 4) * 16;    // B-matrix K half offset
    const float bmask = (lane < 16) ? 1.0f : 0.0f;

    float wsum = 0.0f;

    for (int combo = wave; combo < 25; combo += 8) {
        const int i0 = (combo / 5) * 16;
        const int j0 = (combo % 5) * 16;
        const float* arowP = mp + pbase + (long)(i0 + lrow) * T_LEN;
        const float* browP = mp + pbase + (long)(j0 + lrow) * T_LEN;
        const float* arowT = mt + pbase + (long)(i0 + lrow) * T_LEN;
        const float* browT = mt + pbase + (long)(j0 + lrow) * T_LEN;

        v8f cp = gram_tile(arowP, browP, agrp, bgrp, bmask);
        v8f ct = gram_tile(arowT, browT, agrp, bgrp, bmask);

        #pragma unroll
        for (int e = 0; e < 8; ++e) {
            float d = cp[e] - ct[e];
            wsum += d * d;
        }
    }

    for (int off = 16; off > 0; off >>= 1)
        wsum += __shfl_down(wsum, off);
    if (lane == 0)
        atomicAdd(&acc[A_GRAM], wsum);
}

// ---------------------------------------------------------------------------
// Gate BCE-with-logits: max(x,0) - x*t + log1p(exp(-|x|)), summed.
// ---------------------------------------------------------------------------
__global__ void gate_bce_kernel(const float* __restrict__ x,
                                const float* __restrict__ t,
                                float* __restrict__ acc) {
    float s = 0.0f;
    const int n = B_SZ * T_LEN;
    for (int i = blockIdx.x * blockDim.x + threadIdx.x; i < n;
         i += gridDim.x * blockDim.x) {
        float xv = x[i], tv = t[i];
        s += fmaxf(xv, 0.0f) - xv * tv + log1pf(__expf(-fabsf(xv)));
    }
    __shared__ float red[256];
    red[threadIdx.x] = s;
    __syncthreads();
    for (int sh = 128; sh > 0; sh >>= 1) {
        if (threadIdx.x < sh) red[threadIdx.x] += red[threadIdx.x + sh];
        __syncthreads();
    }
    if (threadIdx.x == 0) atomicAdd(&acc[A_BCE], red[0]);
}

// ---------------------------------------------------------------------------
// Monotonic attention loss: one block per batch; one wave32 per row argmax
// (coalesced, first-max tie-break matching jnp.argmax), peaks in LDS.
// ---------------------------------------------------------------------------
__global__ void mono_kernel(const float* __restrict__ att,
                            float* __restrict__ acc) {
    __shared__ float peaks[T_LEN];
    const int b    = blockIdx.x;
    const int wave = threadIdx.x >> 5;
    const int lane = threadIdx.x & 31;
    const long bbase = (long)b * T_LEN * L_TXT;

    for (int m = wave; m < T_LEN; m += 8) {
        const float* rowp = att + bbase + (long)m * L_TXT;
        float bv = -INFINITY;
        int   bi = 0x7fffffff;
        for (int i = lane; i < L_TXT; i += 32) {
            float v = rowp[i];
            if (v > bv) { bv = v; bi = i; }
        }
        for (int off = 16; off > 0; off >>= 1) {
            float ov = __shfl_down(bv, off);
            int   oi = __shfl_down(bi, off);
            if (ov > bv || (ov == bv && oi < bi)) { bv = ov; bi = oi; }
        }
        if (lane == 0) peaks[m] = (float)bi;
    }
    __syncthreads();

    float s = 0.0f;
    for (int m = threadIdx.x; m < T_LEN - 1; m += blockDim.x)
        s += fmaxf(peaks[m] - peaks[m + 1], 0.0f);

    __shared__ float red[256];
    red[threadIdx.x] = s;
    __syncthreads();
    for (int sh = 128; sh > 0; sh >>= 1) {
        if (threadIdx.x < sh) red[threadIdx.x] += red[threadIdx.x + sh];
        __syncthreads();
    }
    if (threadIdx.x == 0) atomicAdd(&acc[A_MONO], red[0]);
}

// ---------------------------------------------------------------------------
// Finalize: apply mean divisors and loss weights, write the scalar.
// ---------------------------------------------------------------------------
__global__ void finalize_kernel(const float* __restrict__ acc,
                                float* __restrict__ out) {
    if (threadIdx.x == 0 && blockIdx.x == 0) {
        const float NM    = (float)B_SZ * N_MEL * T_LEN;
        const float NS    = (float)B_SZ * N_MEL * (T_LEN - 1);
        const float NBC   = (float)B_SZ * N_MEL;
        const float NG    = (float)B_SZ * T_LEN;
        const float NGRAM = (float)B_SZ * N_MEL * N_MEL;
        const float GSC   = (float)N_MEL * T_LEN;

        float mel        = acc[A_MEL1] / NM + acc[A_MEL2] / NM;
        float gate       = 1.3f * (acc[A_BCE] / NG);
        float spectral   = acc[A_WMSE] / NM + 0.3f * (acc[A_L1] / NM)
                         + 0.2f * (acc[A_SPEC] / NS);
        float perceptual = acc[A_PERC] / NM + 0.5f * (acc[A_L1] / NM);
        float gram       = (acc[A_GRAM] / (GSC * GSC)) / NGRAM;
        float style      = acc[A_MEAN] / NBC + acc[A_VAR] / NBC + 0.5f * gram;
        float mono       = acc[A_MONO] / ((float)B_SZ * T_LEN);

        out[0] = mel + gate + 0.3f * spectral + 0.2f * perceptual
               + 0.1f * style + 0.1f * mono;
    }
}

extern "C" void kernel_launch(void* const* d_in, const int* in_sizes, int n_in,
                              void* d_out, int out_size, void* d_ws, size_t ws_size,
                              hipStream_t stream) {
    (void)in_sizes; (void)n_in; (void)out_size; (void)ws_size;
    const float* mel_out  = (const float*)d_in[0];
    const float* mel_post = (const float*)d_in[1];
    const float* mel_tgt  = (const float*)d_in[2];
    const float* gate_out = (const float*)d_in[3];
    const float* gate_tgt = (const float*)d_in[4];
    const float* att      = (const float*)d_in[5];
    float* acc = (float*)d_ws;
    float* out = (float*)d_out;

    hipLaunchKernelGGL(init_acc_kernel, dim3(1), dim3(32), 0, stream, acc);
    hipLaunchKernelGGL(row_stats_kernel, dim3(B_SZ * N_MEL), dim3(256), 0, stream,
                       mel_out, mel_post, mel_tgt, acc);
    hipLaunchKernelGGL(gram_wmma_kernel, dim3(B_SZ), dim3(256), 0, stream,
                       mel_post, mel_tgt, acc);
    hipLaunchKernelGGL(gate_bce_kernel, dim3(256), dim3(256), 0, stream,
                       gate_out, gate_tgt, acc);
    hipLaunchKernelGGL(mono_kernel, dim3(B_SZ), dim3(256), 0, stream, att, acc);
    hipLaunchKernelGGL(finalize_kernel, dim3(1), dim3(32), 0, stream, acc, out);
}
